// GraphEvolutionDiscr_39900246180399
// MI455X (gfx1250) — matrix operand order repaired and checked by reading, not compiled, not run on hardware
//
#include <hip/hip_runtime.h>
#include <stdint.h>
#include <stddef.h>

// ---------------- problem dims ----------------
#define NN   16384            // nodes
#define EE   262144           // edges (w/o self loops)
#define ET   (EE + NN)        // edges incl. self loops
#define HCH  256              // H*C
#define NH   8                // heads
#define CCH  32               // per-head channels / d_model

typedef __bf16 bf16_t;
typedef __attribute__((ext_vector_type(16))) __bf16 v16bf;
typedef __attribute__((ext_vector_type(8)))  __bf16 v8bf;
typedef __attribute__((ext_vector_type(8)))  float  v8f;

// ---------------- device helpers ----------------
__device__ __forceinline__ unsigned encf(float f) {
  unsigned u = __float_as_uint(f);
  return (u & 0x80000000u) ? ~u : (u | 0x80000000u);   // monotone map for atomicMax
}
__device__ __forceinline__ float decf(unsigned u) {
  return (u & 0x80000000u) ? __uint_as_float(u & 0x7fffffffu) : __uint_as_float(~u);
}
__device__ __forceinline__ float geluf(float x) {
  return 0.5f * x * (1.0f + erff(x * 0.70710678118654752f));
}
__device__ __forceinline__ float lrelu(float m) { return m > 0.f ? m : 0.2f * m; }
__device__ __forceinline__ float wsum32(float v) {
  #pragma unroll
  for (int o = 16; o; o >>= 1) v += __shfl_xor(v, o);
  return v;
}

// ---------------- utility kernels ----------------
__global__ void fill_u32k(unsigned* p, unsigned v, int n) {
  int i = blockIdx.x * blockDim.x + threadIdx.x;
  if (i < n) p[i] = v;
}
__global__ void fill_encnegk(unsigned* p, int n) {
  int i = blockIdx.x * blockDim.x + threadIdx.x;
  if (i < n) p[i] = encf(-3.0e38f);
}
__global__ void build_h0k(const float* __restrict__ x, const float* __restrict__ params,
                          float* __restrict__ h0) {
  int i = blockIdx.x * blockDim.x + threadIdx.x;
  if (i >= NN * 8) return;
  int n = i >> 3, k = i & 7;
  h0[i] = (k < 6) ? x[n * 6 + k] : params[k - 6];
}
__global__ void esumk(const float* __restrict__ ea, float* __restrict__ s) {
  __shared__ float red[256];
  int i = blockIdx.x * blockDim.x + threadIdx.x;
  red[threadIdx.x] = (i < EE) ? ea[i] : 0.f;
  __syncthreads();
  for (int o = 128; o; o >>= 1) {
    if (threadIdx.x < o) red[threadIdx.x] += red[threadIdx.x + o];
    __syncthreads();
  }
  if (threadIdx.x == 0) atomicAdd(s, red[0]);
}
__global__ void cvt_bf16k(const float* __restrict__ s, bf16_t* __restrict__ d, int n) {
  int i = blockIdx.x * blockDim.x + threadIdx.x;
  if (i < n) d[i] = (bf16_t)s[i];
}

// Pack four 256x256 fp32 weight matrices into WMMA B-fragment order (bf16).
// B tile = 32(K) x 16(N).  lane = 16*(Khalf) + n%16 ; element j -> K = kt*32 + 16*(lane>>4) + j
__global__ void pack_wbk(const float* __restrict__ W0, const float* __restrict__ W1,
                         const float* __restrict__ W2, const float* __restrict__ W3,
                         bf16_t* __restrict__ Wp) {
  int idx = blockIdx.x * blockDim.x + threadIdx.x;        // 4 * 65536
  if (idx >= 4 * 65536) return;
  int mi = idx >> 16, r = idx & 65535;
  int j = r & 15, lane = (r >> 4) & 31, nt = (r >> 9) & 15, kt = r >> 13;
  int n = nt * 16 + (lane & 15);
  int k = kt * 32 + ((lane >> 4) << 4) + j;
  const float* W = (mi == 0) ? W0 : (mi == 1) ? W1 : (mi == 2) ? W2 : W3;
  Wp[idx] = (bf16_t)W[k * 256 + n];
}

// g0 dense: fin = 8 -> plain VALU (K too small for WMMA to pay)
__global__ void g0_densek(const float* __restrict__ h0,
                          const float* __restrict__ Wl, const float* __restrict__ bl,
                          const float* __restrict__ Wr, const float* __restrict__ br,
                          float* __restrict__ xl, float* __restrict__ xr) {
  int i = blockIdx.x * blockDim.x + threadIdx.x;
  if (i >= NN * HCH) return;
  int n = i >> 8, c = i & 255;
  float l = bl[c], r = br[c];
  #pragma unroll
  for (int k = 0; k < 8; ++k) {
    float hv = h0[n * 8 + k];
    l += hv * Wl[k * 256 + c];
    r += hv * Wr[k * 256 + c];
  }
  xl[i] = l; xr[i] = r;
}

// ---------------- WMMA GEMM: C[N x 256] = A[N x 256] * B + bias ----------------
// 128 threads = 4 waves per block; each wave owns 64 output columns; grid.x = NN/16
__global__ void gemm_n256k(const bf16_t* __restrict__ A, const bf16_t* __restrict__ Bp,
                           const float* __restrict__ bias, float* __restrict__ C) {
  const int lane = threadIdx.x & 31;
  const int wv   = threadIdx.x >> 5;
  const int rt   = blockIdx.x;
  const int arow = rt * 16 + (lane & 15);
  const int kh   = (lane >> 4) * 16;         // A layout: K 0-15 / 16-31 per lane half
  v8f acc[4] = {};
  #pragma unroll
  for (int kt = 0; kt < 8; ++kt) {
    union { v16bf v; v8bf h[2]; } a, b0, b1, b2, b3;
    const bf16_t* ap = A + arow * 256 + kt * 32 + kh;
    a.h[0] = *(const v8bf*)(ap);             // K = base .. base+7
    a.h[1] = *(const v8bf*)(ap + 8);         // K = base+8 .. base+15
    const bf16_t* bp = Bp + ((kt * 16 + wv * 4) * 32 + lane) * 16;
    b0.h[0] = *(const v8bf*)(bp);        b0.h[1] = *(const v8bf*)(bp + 8);
    b1.h[0] = *(const v8bf*)(bp + 512);  b1.h[1] = *(const v8bf*)(bp + 520);
    b2.h[0] = *(const v8bf*)(bp + 1024); b2.h[1] = *(const v8bf*)(bp + 1032);
    b3.h[0] = *(const v8bf*)(bp + 1536); b3.h[1] = *(const v8bf*)(bp + 1544);
    acc[0] = __builtin_amdgcn_wmma_f32_16x16x32_bf16(false, a.v, false, b0.v, (short)0, acc[0], false, false);
    acc[1] = __builtin_amdgcn_wmma_f32_16x16x32_bf16(false, a.v, false, b1.v, (short)0, acc[1], false, false);
    acc[2] = __builtin_amdgcn_wmma_f32_16x16x32_bf16(false, a.v, false, b2.v, (short)0, acc[2], false, false);
    acc[3] = __builtin_amdgcn_wmma_f32_16x16x32_bf16(false, a.v, false, b3.v, (short)0, acc[3], false, false);
  }
  const int rbase = rt * 16 + 8 * (lane >> 4);    // C/D: VGPR v -> M = v + 8*(lane/16)
  #pragma unroll
  for (int s = 0; s < 4; ++s) {
    const int col = wv * 64 + s * 16 + (lane & 15);
    const float bv = bias[col];
    #pragma unroll
    for (int v = 0; v < 8; ++v)
      C[(rbase + v) * 256 + col] = acc[s][v] + bv;
  }
}

// ---------------- edge softmax passes (one thread per (edge, head)) ----------------
__global__ void edge_pass1k(const float* __restrict__ xl, const float* __restrict__ xr,
                            const int* __restrict__ ei0, const int* __restrict__ ei1,
                            const float* __restrict__ att,
                            const float* __restrict__ eattr, const float* __restrict__ esum,
                            const float* __restrict__ We,
                            float* __restrict__ abuf, unsigned* __restrict__ amax) {
  int idx = blockIdx.x * blockDim.x + threadIdx.x;
  if (idx >= ET * NH) return;
  int e = idx >> 3, hd = idx & 7;
  int s = (e < EE) ? ei0[e] : (e - EE);
  int d = (e < EE) ? ei1[e] : (e - EE);
  const float4* pl = (const float4*)(xl + s * HCH + hd * 32);
  const float4* pr = (const float4*)(xr + d * HCH + hd * 32);
  const float4* pa = (const float4*)(att + hd * 32);
  const bool useE = (We != nullptr);
  float eav = 0.f;
  const float4* pe = nullptr;
  if (useE) {
    eav = (e < EE) ? eattr[e] : esum[0] * (1.0f / EE);
    pe = (const float4*)(We + hd * 32);
  }
  float a = 0.f;
  #pragma unroll
  for (int q = 0; q < 8; ++q) {
    float4 L = pl[q], R = pr[q], T = pa[q];
    float mx = L.x + R.x, my = L.y + R.y, mz = L.z + R.z, mw = L.w + R.w;
    if (useE) { float4 Ev = pe[q]; mx += eav * Ev.x; my += eav * Ev.y; mz += eav * Ev.z; mw += eav * Ev.w; }
    a += lrelu(mx) * T.x + lrelu(my) * T.y + lrelu(mz) * T.z + lrelu(mw) * T.w;
  }
  abuf[idx] = a;
  atomicMax(amax + d * NH + hd, encf(a));
}

__global__ void edge_pass2k(const int* __restrict__ ei1, const unsigned* __restrict__ amax,
                            float* __restrict__ abuf, float* __restrict__ den) {
  int idx = blockIdx.x * blockDim.x + threadIdx.x;
  if (idx >= ET * NH) return;
  int e = idx >> 3, hd = idx & 7;
  int d = (e < EE) ? ei1[e] : (e - EE);
  float ex = expf(abuf[idx] - decf(amax[d * NH + hd]));
  abuf[idx] = ex;
  atomicAdd(den + d * NH + hd, ex);
}

__global__ void edge_pass3k(const float* __restrict__ xl,
                            const int* __restrict__ ei0, const int* __restrict__ ei1,
                            const float* __restrict__ abuf, const float* __restrict__ den,
                            float* __restrict__ acc) {
  int idx = blockIdx.x * blockDim.x + threadIdx.x;
  if (idx >= ET * NH) return;
  int e = idx >> 3, hd = idx & 7;
  int s = (e < EE) ? ei0[e] : (e - EE);
  int d = (e < EE) ? ei1[e] : (e - EE);
  float alpha = abuf[idx] / den[d * NH + hd];
  const float4* pl = (const float4*)(xl + s * HCH + hd * 32);
  float* po = acc + d * HCH + hd * 32;
  #pragma unroll
  for (int q = 0; q < 8; ++q) {
    float4 L = pl[q];
    atomicAdd(po + q * 4 + 0, alpha * L.x);
    atomicAdd(po + q * 4 + 1, alpha * L.y);
    atomicAdd(po + q * 4 + 2, alpha * L.z);
    atomicAdd(po + q * 4 + 3, alpha * L.w);
  }
}

__global__ void fin_concatk(const float* __restrict__ acc, const float* __restrict__ bias,
                            float* __restrict__ dst) {
  int i = blockIdx.x * blockDim.x + threadIdx.x;
  if (i >= NN * HCH) return;
  dst[i] = geluf(acc[i] + bias[i & 255]);
}

__global__ void fin_meank(const float* __restrict__ acc, const float* __restrict__ bias,
                          float* __restrict__ t) {
  int i = blockIdx.x * blockDim.x + threadIdx.x;
  if (i >= NN * CCH) return;
  int n = i >> 5, c = i & 31;
  float s = 0.f;
  #pragma unroll
  for (int h = 0; h < NH; ++h) s += acc[n * HCH + h * 32 + c];
  t[i] = geluf(s * 0.125f + bias[c]);
}

// ---------------- fused transformer decoder: one wave32 per node, lane = feature ----------------
__device__ __forceinline__ float lnf(float x, const float* g, const float* b, int lane) {
  float m = wsum32(x) * (1.0f / 32.0f);
  float dd = x - m;
  float var = wsum32(dd * dd) * (1.0f / 32.0f);
  return dd * rsqrtf(var + 1e-5f) * g[lane] + b[lane];
}

__global__ void decoderk(const float* __restrict__ tbuf,
                         const float* Wv_sa, const float* bv_sa, const float* Wo_sa, const float* bo_sa,
                         const float* Wv_ca, const float* bv_ca, const float* Wo_ca, const float* bo_ca,
                         const float* W1, const float* b1f, const float* W2, const float* b2f,
                         const float* ln1g, const float* ln1b, const float* ln2g, const float* ln2b,
                         const float* ln3g, const float* ln3b,
                         const float* convW, const float* convb,
                         float* __restrict__ out) {
  int lane = threadIdx.x & 31;
  int node = (blockIdx.x * blockDim.x + threadIdx.x) >> 5;
  if (node >= NN) return;
  float t = tbuf[node * 32 + lane];
  const float mem = t;
  #pragma unroll
  for (int l = 0; l < 2; ++l) {
    // self-attn branch (seq-len 1 => out_proj(v_proj(t)))
    float u = bv_sa[l * 32 + lane];
    for (int k = 0; k < 32; ++k) u += __shfl(t, k) * Wv_sa[l * 1024 + k * 32 + lane];
    float sa = bo_sa[l * 32 + lane];
    for (int k = 0; k < 32; ++k) sa += __shfl(u, k) * Wo_sa[l * 1024 + k * 32 + lane];
    t = lnf(t + sa, ln1g + l * 32, ln1b + l * 32, lane);
    // cross-attn branch from mem
    float v = bv_ca[l * 32 + lane];
    for (int k = 0; k < 32; ++k) v += __shfl(mem, k) * Wv_ca[l * 1024 + k * 32 + lane];
    float ca = bo_ca[l * 32 + lane];
    for (int k = 0; k < 32; ++k) ca += __shfl(v, k) * Wo_ca[l * 1024 + k * 32 + lane];
    t = lnf(t + ca, ln2g + l * 32, ln2b + l * 32, lane);
    // FFN 32 -> 128 -> 32
    float f0 = b1f[l * 128 + lane], f1 = b1f[l * 128 + 32 + lane];
    float f2 = b1f[l * 128 + 64 + lane], f3 = b1f[l * 128 + 96 + lane];
    for (int k = 0; k < 32; ++k) {
      float tk = __shfl(t, k);
      const float* w = W1 + l * 4096 + k * 128;
      f0 += tk * w[lane]; f1 += tk * w[32 + lane]; f2 += tk * w[64 + lane]; f3 += tk * w[96 + lane];
    }
    f0 = fmaxf(f0, 0.f); f1 = fmaxf(f1, 0.f); f2 = fmaxf(f2, 0.f); f3 = fmaxf(f3, 0.f);
    float o = b2f[l * 32 + lane];
    for (int k = 0; k < 32; ++k) {
      o += __shfl(f0, k) * W2[l * 4096 + k * 32 + lane];
      o += __shfl(f1, k) * W2[l * 4096 + (32 + k) * 32 + lane];
      o += __shfl(f2, k) * W2[l * 4096 + (64 + k) * 32 + lane];
      o += __shfl(f3, k) * W2[l * 4096 + (96 + k) * 32 + lane];
    }
    t = lnf(t + o, ln3g + l * 32, ln3b + l * 32, lane);
  }
  float y = tanhf(t);
  float o0 = 0.f, o1 = 0.f, o2 = 0.f, o3 = 0.f;
  for (int k = 0; k < 32; ++k) {
    float yk = __shfl(y, k);
    const float* w = convW + k * 4;
    o0 += yk * w[0]; o1 += yk * w[1]; o2 += yk * w[2]; o3 += yk * w[3];
  }
  if (lane == 0) {
    float* op = out + node * 4;
    op[0] = o0 + convb[0]; op[1] = o1 + convb[1]; op[2] = o2 + convb[2]; op[3] = o3 + convb[3];
  }
}

// ---------------- launcher ----------------
extern "C" void kernel_launch(void* const* d_in, const int* in_sizes, int n_in,
                              void* d_out, int out_size, void* d_ws, size_t ws_size,
                              hipStream_t stream) {
  (void)in_sizes; (void)n_in; (void)out_size; (void)ws_size;
  const float* x      = (const float*)d_in[0];
  const int*   ei     = (const int*)d_in[1];
  const float* eattr  = (const float*)d_in[2];
  const float* params = (const float*)d_in[3];
  const float* g0_Wl  = (const float*)d_in[4];
  const float* g0_bl  = (const float*)d_in[5];
  const float* g0_Wr  = (const float*)d_in[6];
  const float* g0_br  = (const float*)d_in[7];
  const float* g0_att = (const float*)d_in[8];
  const float* g0_We  = (const float*)d_in[9];
  const float* g0_bias= (const float*)d_in[10];
  const float* g1_Wl  = (const float*)d_in[11];
  const float* g1_bl  = (const float*)d_in[12];
  const float* g1_Wr  = (const float*)d_in[13];
  const float* g1_br  = (const float*)d_in[14];
  const float* g1_att = (const float*)d_in[15];
  const float* g1_bias= (const float*)d_in[16];
  const float* g2_Wl  = (const float*)d_in[17];
  const float* g2_bl  = (const float*)d_in[18];
  const float* g2_Wr  = (const float*)d_in[19];
  const float* g2_br  = (const float*)d_in[20];
  const float* g2_att = (const float*)d_in[21];
  const float* g2_bias= (const float*)d_in[22];
  const float* Wv_sa  = (const float*)d_in[23];
  const float* bv_sa  = (const float*)d_in[24];
  const float* Wo_sa  = (const float*)d_in[25];
  const float* bo_sa  = (const float*)d_in[26];
  const float* Wv_ca  = (const float*)d_in[27];
  const float* bv_ca  = (const float*)d_in[28];
  const float* Wo_ca  = (const float*)d_in[29];
  const float* bo_ca  = (const float*)d_in[30];
  const float* W1     = (const float*)d_in[31];
  const float* b1     = (const float*)d_in[32];
  const float* W2     = (const float*)d_in[33];
  const float* b2     = (const float*)d_in[34];
  const float* ln1g   = (const float*)d_in[35];
  const float* ln1b   = (const float*)d_in[36];
  const float* ln2g   = (const float*)d_in[37];
  const float* ln2b   = (const float*)d_in[38];
  const float* ln3g   = (const float*)d_in[39];
  const float* ln3b   = (const float*)d_in[40];
  const float* convW  = (const float*)d_in[41];
  const float* convb  = (const float*)d_in[42];
  const int* ei0 = ei;
  const int* ei1 = ei + EE;
  float* out = (float*)d_out;

  // ---- workspace layout (all chunks 256B-aligned) ----
  char* ws = (char*)d_ws;
  size_t off = 0;
  float*   hA   = (float*)(ws + off);  off += (size_t)NN * HCH * 4;   // 16 MB
  float*   hB   = (float*)(ws + off);  off += (size_t)NN * HCH * 4;   // 16 MB
  bf16_t*  hb   = (bf16_t*)(ws + off); off += (size_t)NN * HCH * 2;   //  8 MB
  float*   xl   = (float*)(ws + off);  off += (size_t)NN * HCH * 4;   // 16 MB
  float*   xr   = (float*)(ws + off);  off += (size_t)NN * HCH * 4;   // 16 MB
  float*   abuf = (float*)(ws + off);  off += (size_t)ET * NH * 4;    // ~8.9 MB
  unsigned* amax= (unsigned*)(ws + off); off += (size_t)NN * NH * 4;  // 0.5 MB
  float*   den  = (float*)(ws + off);  off += (size_t)NN * NH * 4;    // 0.5 MB
  bf16_t*  wp   = (bf16_t*)(ws + off); off += (size_t)4 * 65536 * 2;  // 0.5 MB
  float*   h0   = (float*)(ws + off);  off += (size_t)NN * 8 * 4;     // 0.5 MB
  float*   tb   = (float*)(ws + off);  off += (size_t)NN * CCH * 4;   //  2 MB
  float*   ssum = (float*)(ws + off);  off += 256;

  const int B = 256;
  const int grNode256 = (NN * HCH) / B;   // 16384
  const int grNode8   = (NN * 8) / B;     // 512
  const int grE       = (ET * NH + B - 1) / B; // 8704

  // ---- prep ----
  fill_u32k<<<1, B, 0, stream>>>((unsigned*)ssum, 0u, 1);
  build_h0k<<<grNode8, B, 0, stream>>>(x, params, h0);
  esumk<<<EE / B, B, 0, stream>>>(eattr, ssum);
  pack_wbk<<<(4 * 65536) / B, B, 0, stream>>>(g1_Wl, g1_Wr, g2_Wl, g2_Wr, wp);

  // ================= g0 (fin=8, edge features) =================
  g0_densek<<<grNode256, B, 0, stream>>>(h0, g0_Wl, g0_bl, g0_Wr, g0_br, xl, xr);
  fill_u32k<<<grNode256, B, 0, stream>>>((unsigned*)hB, 0u, NN * HCH);
  fill_u32k<<<grNode8, B, 0, stream>>>((unsigned*)den, 0u, NN * NH);
  fill_encnegk<<<grNode8, B, 0, stream>>>(amax, NN * NH);
  edge_pass1k<<<grE, B, 0, stream>>>(xl, xr, ei0, ei1, g0_att, eattr, ssum, g0_We, abuf, amax);
  edge_pass2k<<<grE, B, 0, stream>>>(ei1, amax, abuf, den);
  edge_pass3k<<<grE, B, 0, stream>>>(xl, ei0, ei1, abuf, den, hB);
  fin_concatk<<<grNode256, B, 0, stream>>>(hB, g0_bias, hA);

  // ================= g1 (256 -> 256, WMMA) =================
  cvt_bf16k<<<grNode256, B, 0, stream>>>(hA, hb, NN * HCH);
  gemm_n256k<<<NN / 16, 128, 0, stream>>>(hb, wp,          g1_bl, xl);
  gemm_n256k<<<NN / 16, 128, 0, stream>>>(hb, wp + 65536,  g1_br, xr);
  fill_u32k<<<grNode256, B, 0, stream>>>((unsigned*)hB, 0u, NN * HCH);
  fill_u32k<<<grNode8, B, 0, stream>>>((unsigned*)den, 0u, NN * NH);
  fill_encnegk<<<grNode8, B, 0, stream>>>(amax, NN * NH);
  edge_pass1k<<<grE, B, 0, stream>>>(xl, xr, ei0, ei1, g1_att, eattr, ssum, nullptr, abuf, amax);
  edge_pass2k<<<grE, B, 0, stream>>>(ei1, amax, abuf, den);
  edge_pass3k<<<grE, B, 0, stream>>>(xl, ei0, ei1, abuf, den, hB);
  fin_concatk<<<grNode256, B, 0, stream>>>(hB, g1_bias, hB);

  // ================= g2 (256 -> 256, WMMA, concat=False) =================
  cvt_bf16k<<<grNode256, B, 0, stream>>>(hB, hb, NN * HCH);
  gemm_n256k<<<NN / 16, 128, 0, stream>>>(hb, wp + 2 * 65536, g2_bl, xl);
  gemm_n256k<<<NN / 16, 128, 0, stream>>>(hb, wp + 3 * 65536, g2_br, xr);
  fill_u32k<<<grNode256, B, 0, stream>>>((unsigned*)hA, 0u, NN * HCH);
  fill_u32k<<<grNode8, B, 0, stream>>>((unsigned*)den, 0u, NN * NH);
  fill_encnegk<<<grNode8, B, 0, stream>>>(amax, NN * NH);
  edge_pass1k<<<grE, B, 0, stream>>>(xl, xr, ei0, ei1, g2_att, eattr, ssum, nullptr, abuf, amax);
  edge_pass2k<<<grE, B, 0, stream>>>(ei1, amax, abuf, den);
  edge_pass3k<<<grE, B, 0, stream>>>(xl, ei0, ei1, abuf, den, hA);
  fin_meank<<<(NN * CCH) / B, B, 0, stream>>>(hA, g2_bias, tb);

  // ================= fused decoder + output conv =================
  decoderk<<<(NN * 32) / B, B, 0, stream>>>(tb,
      Wv_sa, bv_sa, Wo_sa, bo_sa, Wv_ca, bv_ca, Wo_ca, bo_ca,
      W1, b1, W2, b2, ln1g, ln1b, ln2g, ln2b, ln3g, ln3b,
      convW, convb, out);
}
